// TopkRouting_16569983828344
// MI455X (gfx1250) — matrix operand order repaired and checked by reading, not compiled, not run on hardware
//
#include <hip/hip_runtime.h>

// ---------- types ----------
typedef __bf16  bf16_t;
typedef __attribute__((ext_vector_type(8)))  __bf16 v8bf;
typedef __attribute__((ext_vector_type(16))) __bf16 v16bf;
typedef __attribute__((ext_vector_type(8)))  float  v8f;

#define QD        256
#define NROWS     65536          // B*N = 64*1024
#define QK_ELEMS  16777216       // 65536*256
#define W_ELEMS   65536          // 256*256
#define OUT_HALF  1048576        // 64*1024*16
#define NEG_INF   (-3.402823466e38f)

// concat two 8-wide bf16 vectors into the 16-wide WMMA operand
__device__ __forceinline__ v16bf cat8(v8bf lo, v8bf hi) {
  v16bf r;
#pragma unroll
  for (int i = 0; i < 8; ++i) { r[i] = lo[i]; r[i + 8] = hi[i]; }
  return r;
}

// convert 8 floats @p0 and 8 floats @p1 into one v16bf fragment
__device__ __forceinline__ v16bf cvt16(const float* p0, const float* p1) {
  float4 f0 = ((const float4*)p0)[0];
  float4 f1 = ((const float4*)p0)[1];
  float4 f2 = ((const float4*)p1)[0];
  float4 f3 = ((const float4*)p1)[1];
  v16bf r;
  r[0]  = (__bf16)f0.x; r[1]  = (__bf16)f0.y; r[2]  = (__bf16)f0.z; r[3]  = (__bf16)f0.w;
  r[4]  = (__bf16)f1.x; r[5]  = (__bf16)f1.y; r[6]  = (__bf16)f1.z; r[7]  = (__bf16)f1.w;
  r[8]  = (__bf16)f2.x; r[9]  = (__bf16)f2.y; r[10] = (__bf16)f2.z; r[11] = (__bf16)f2.w;
  r[12] = (__bf16)f3.x; r[13] = (__bf16)f3.y; r[14] = (__bf16)f3.z; r[15] = (__bf16)f3.w;
  return r;
}

// ================= Phase 0: one-shot weight f32 -> bf16 conversion ===========
// wq/wk are 256KB each; convert once so the projection loop loads bf16 B tiles
// directly (no per-WMMA cvt on the B side, half the weight bytes).
__global__ __launch_bounds__(256)
void wcvt_kernel(const float* __restrict__ wq, const float* __restrict__ wk,
                 unsigned short* __restrict__ ws) {
  bf16_t* dq = (bf16_t*)ws + 2 * (size_t)QK_ELEMS;
  bf16_t* dk = dq + W_ELEMS;
  const int i = (blockIdx.x * 256 + threadIdx.x) * 4;   // grid 128 -> 131072 elems
  const float* src; bf16_t* dst; int j;
  if (i < W_ELEMS) { src = wq; dst = dq; j = i; }
  else             { src = wk; dst = dk; j = i - W_ELEMS; }
  float4 f = *(const float4*)(src + j);
  dst[j]     = (bf16_t)f.x;
  dst[j + 1] = (bf16_t)f.y;
  dst[j + 2] = (bf16_t)f.z;
  dst[j + 3] = (bf16_t)f.w;
}

// ======================= Phase 1: fused q/k projection =======================
// One wave -> one 16-row tile x EIGHT 16-col e-tiles. The 8 A fragments
// (g_win rows, converted f32->bf16 once) stay in VGPRs across all 8 e-tiles,
// so A-side cvt cost is amortized 8x: ~1 cvt + 2 bf16 loads per WMMA.
__global__ __launch_bounds__(256)
void proj_kernel(const float* __restrict__ x,
                 const float* __restrict__ bq, const float* __restrict__ bk,
                 unsigned short* __restrict__ ws) {
  const int lane = threadIdx.x & 31;
  const int wave = threadIdx.x >> 5;
  const int t    = blockIdx.x * 8 + wave;      // 0 .. 16383
  const bool is_k = (t >= 8192);               // 8192 wave-jobs per matrix
  const int tt      = t & 8191;
  const int rowtile = tt >> 1;                 // 0 .. 4095
  const int ehalf   = (tt & 1) * 128;          // e-base of this wave's 8 tiles

  const bf16_t* wb = (const bf16_t*)ws + 2 * (size_t)QK_ELEMS + (is_k ? W_ELEMS : 0);
  const float*  Bv = is_k ? bk : bq;
  bf16_t* dst = (bf16_t*)ws + (is_k ? (size_t)QK_ELEMS : 0);

  // ---- A fragments: convert this tile's 16 g_win rows once ----
  const int arow = rowtile * 16 + (lane & 15);
  const int aoff = (lane < 16) ? 0 : 8;        // ISA 16-bit A-fragment layout
  const float* ap = x + (size_t)arow * QD;
  v16bf afrag[8];
#pragma unroll
  for (int kk = 0; kk < 8; ++kk)
    afrag[kk] = cvt16(ap + kk * 32 + aoff, ap + kk * 32 + aoff + 16);

  const int koff  = (lane < 16) ? 0 : 8;       // B-fragment K split across lane halves
  const int rbase = rowtile * 16 + ((lane < 16) ? 0 : 8);
  const float scale = is_k ? 1.0f : 0.0625f;   // fold 256^-0.5 into q

  for (int et = 0; et < 8; ++et) {
    const int ecol = ehalf + et * 16 + (lane & 15);
    const bf16_t* wp = wb + (size_t)ecol * QD;
    v8f acc = {};
#pragma unroll
    for (int kk = 0; kk < 8; ++kk) {
      const bf16_t* p = wp + kk * 32 + koff;
      v16bf bfrag = cat8(*(const v8bf*)p, *(const v8bf*)(p + 16));
      acc = __builtin_amdgcn_wmma_f32_16x16x32_bf16(
          false, afrag[kk], false, bfrag, (short)0, acc, false, false);
    }
    const float bias = Bv[ecol];
#pragma unroll
    for (int v = 0; v < 8; ++v) {
      float val = (acc[v] + bias) * scale;
      dst[(size_t)(rbase + v) * QD + ecol] = (bf16_t)val;   // coalesced per-row
    }
  }
}

// ================= Phase 2: fused logits + top-16 + softmax ==================
// One workgroup (16 waves) per (batch, 16-row tile). Logits block 16x1024 lives
// in LDS only (never hits HBM). Wave w computes column tiles 4w..4w+3, then does
// iterative top-16 + softmax for row w.
__global__ __launch_bounds__(512)
void route_kernel(const unsigned short* __restrict__ ws,
                  float* __restrict__ out) {
  __shared__ float s_logits[16 * 1024];        // 64 KB

  const int lane = threadIdx.x & 31;
  const int wave = threadIdx.x >> 5;           // 0..15
  const int b  = blockIdx.x >> 6;              // batch
  const int rt = blockIdx.x & 63;              // row tile

  const bf16_t* q = ((const bf16_t*)ws);
  const bf16_t* k = ((const bf16_t*)ws) + QK_ELEMS;

  // ---- preload the 8 A fragments (q rows of this tile) ----
  const int aoff = (lane < 16) ? 0 : 8;
  const bf16_t* ap = q + ((size_t)b * 1024 + rt * 16 + (lane & 15)) * QD;
  v16bf afrag[8];
#pragma unroll
  for (int kk = 0; kk < 8; ++kk) {
    const bf16_t* p = ap + kk * 32 + aoff;
    afrag[kk] = cat8(*(const v8bf*)p, *(const v8bf*)(p + 16));
  }

  // ---- logits: 4 column tiles per wave ----
  const int koff = (lane < 16) ? 0 : 8;
#pragma unroll
  for (int cj = 0; cj < 4; ++cj) {
    const int ct = wave * 4 + cj;
    const bf16_t* bp = k + ((size_t)b * 1024 + ct * 16 + (lane & 15)) * QD;
    v8f acc = {};
#pragma unroll
    for (int kk = 0; kk < 8; ++kk) {
      const bf16_t* p = bp + kk * 32 + koff;
      v16bf bfrag = cat8(*(const v8bf*)p, *(const v8bf*)(p + 16));
      acc = __builtin_amdgcn_wmma_f32_16x16x32_bf16(
          false, afrag[kk], false, bfrag, (short)0, acc, false, false);
    }
    const int col  = ct * 16 + (lane & 15);
    const int mrow = (lane < 16) ? 0 : 8;
#pragma unroll
    for (int v = 0; v < 8; ++v)
      s_logits[(mrow + v) * 1024 + col] = acc[v];
  }
  __syncthreads();

  // ---- top-16 for row `wave`: 32 candidates per lane, in registers ----
  float vv[32];
#pragma unroll
  for (int j = 0; j < 32; ++j)
    vv[j] = s_logits[wave * 1024 + j * 32 + lane];   // stride-1 across lanes

  float sel_v = NEG_INF;
  int   sel_i = 0;
#pragma unroll 1
  for (int t = 0; t < 16; ++t) {
    // local argmax over this lane's 32 survivors
    float bv = NEG_INF; int bj = 0;
#pragma unroll
    for (int j = 0; j < 32; ++j)
      if (vv[j] > bv) { bv = vv[j]; bj = j; }
    int bc = bj * 32 + lane;
    // wave-wide argmax (ties -> lowest column, matching top_k order)
#pragma unroll
    for (int off = 16; off > 0; off >>= 1) {
      float ov = __shfl_xor(bv, off, 32);
      int   oc = __shfl_xor(bc, off, 32);
      if (ov > bv || (ov == bv && oc < bc)) { bv = ov; bc = oc; }
    }
    if (lane == t) { sel_v = bv; sel_i = bc; }
    // winner's owner lane knocks it out
    const int  jj   = bc >> 5;
    const bool mine = ((bc & 31) == lane);
#pragma unroll
    for (int j = 0; j < 32; ++j)
      if (mine && (jj == j)) vv[j] = NEG_INF;
  }

  // ---- softmax over the 16 selected logits (lanes 0..15 hold t=lane) ----
  const float m = __shfl(sel_v, 0, 32);        // t=0 is the max
  const float e = __expf(sel_v - m);
  float s = e;
#pragma unroll
  for (int off = 8; off > 0; off >>= 1)        // reduce within 16-lane group
    s += __shfl_xor(s, off, 32);
  const float wt = e / s;

  if (lane < 16) {
    const size_t o = ((size_t)b * 1024 + rt * 16 + wave) * 16 + lane;
    out[o]            = wt;                    // r_weight
    out[OUT_HALF + o] = (float)sel_i;          // topk_index (exact in f32)
  }
}

// =============================== launcher ====================================
extern "C" void kernel_launch(void* const* d_in, const int* in_sizes, int n_in,
                              void* d_out, int out_size, void* d_ws, size_t ws_size,
                              hipStream_t stream) {
  (void)in_sizes; (void)n_in; (void)out_size; (void)ws_size;
  const float* g_win = (const float*)d_in[0];
  const float* wq    = (const float*)d_in[1];
  const float* bq    = (const float*)d_in[2];
  const float* wk    = (const float*)d_in[3];
  const float* bk    = (const float*)d_in[4];
  unsigned short* ws = (unsigned short*)d_ws;
  // ws layout (bf16 elems): [0,16M) q | [16M,32M) k | [32M,32M+64K) wq | +64K wk

  // one-shot weight conversion: 131072 elems, 4 per thread
  wcvt_kernel<<<128, 256, 0, stream>>>(wq, wk, ws);
  // 16384 wave-jobs (rowtile x half x {q,k}), 8 waves per 256-thread block
  proj_kernel<<<2048, 256, 0, stream>>>(g_win, bq, bk, ws);
  // one workgroup per (batch, 16-row tile): 64*64 = 4096 blocks, 16 waves each
  route_kernel<<<4096, 512, 0, stream>>>(ws, (float*)d_out);
}
